// AllAtomPotts_65335042506827
// MI455X (gfx1250) — matrix operand order repaired and checked by reading, not compiled, not run on hardware
//
#include <hip/hip_runtime.h>
#include <math.h>

// ---------------------------------------------------------------------------
// Types for CDNA5 WMMA (wave32): v_wmma_f32_16x16x32_bf16
// ---------------------------------------------------------------------------
typedef __bf16 bf16;
typedef __attribute__((ext_vector_type(16))) __bf16 v16bf;
typedef __attribute__((ext_vector_type(8)))  float  v8f;
typedef __attribute__((ext_vector_type(4)))  float  v4f;

#define N_RES 1536
#define NATM  14
#define KNB   48
#define KAA   32
#define DM    128
#define NKP   (N_RES*KNB)   // 73728
#define BIGF  3.0e38f

__device__ inline float gelu_t(float x){
  float x3 = x*x*x;
  return 0.5f*x*(1.0f + tanhf(0.7978845608028654f*(x + 0.044715f*x3)));
}
__device__ inline float sigm(float x){ return 1.0f/(1.0f+expf(-x)); }

// -------------------- reductions ------------------------------------------
__device__ inline float waveSum(float x){
  #pragma unroll
  for(int o=16;o>0;o>>=1) x += __shfl_xor(x,o,32);
  return x;
}
__device__ inline float blkSum128(float v, float* sb){
  int t=threadIdx.x; sb[t]=v; __syncthreads();
  for(int s=64;s>0;s>>=1){ if(t<s) sb[t]+=sb[t+s]; __syncthreads(); }
  float r=sb[0]; __syncthreads(); return r;
}
__device__ inline float blkMax128(float v, float* sb){
  int t=threadIdx.x; sb[t]=v; __syncthreads();
  for(int s=64;s>0;s>>=1){ if(t<s) sb[t]=fmaxf(sb[t],sb[t+s]); __syncthreads(); }
  float r=sb[0]; __syncthreads(); return r;
}

// ---------------------------------------------------------------------------
// Weight pre-swizzle: f32 [Kc x Ncw] row-major  ->  per-lane B fragments.
// Layout: Bp[(((ntile*nKT)+kt)*32 + lane)*16 + h] = W[kt*32 + (lane>>4)*16 + h][ntile*16 + (lane&15)]
// so a wave's B fragment for a k-tile is 2 contiguous 16B loads per lane.
// ---------------------------------------------------------------------------
__global__ void conv_weight_kernel(const float* __restrict__ W, bf16* __restrict__ Bp,
                                   int Kc, int Ncw, int NcPad){
  int idx = blockIdx.x*256 + threadIdx.x;
  int total = Kc*NcPad;
  if(idx >= total) return;
  int perTile = Kc*16;              // elements per n-tile (= nKT*512)
  int ntile = idx / perTile;
  int rem   = idx % perTile;
  int kt    = rem / 512;
  int r2    = rem % 512;
  int lane  = r2 / 16;
  int h     = r2 % 16;
  int nn = ntile*16 + (lane & 15);
  int k  = kt*32 + (lane >> 4)*16 + h;
  float val = (nn < Ncw) ? W[(size_t)k*Ncw + nn] : 0.0f;
  Bp[idx] = (bf16)val;
}

// ---------------------------------------------------------------------------
// Generic bf16 WMMA GEMM: C[M,NcOut] = act((A[M,Kc] @ B + bias) * scale)
// block = 256 threads = 8 waves, each wave one 16x16 tile; grid (M/128, NcPad/16)
// act: 0=none 1=gelu(tanh) 2=silu 3=sigmoid
// ---------------------------------------------------------------------------
union FragBF { v16bf v; v4f q[2]; };

__global__ void gemm_wmma_kernel(const bf16* __restrict__ A, const bf16* __restrict__ Bp,
                                 float* __restrict__ outF, bf16* __restrict__ outB,
                                 const float* __restrict__ bias, const float* __restrict__ scaleP,
                                 int M, int Kc, int NcOut, int act){
  int wave = threadIdx.x >> 5;
  int lane = threadIdx.x & 31;
  int m0 = (blockIdx.x*8 + wave)*16;
  int n0 = blockIdx.y*16;
  int half = lane >> 4, l15 = lane & 15;
  int nKT = Kc >> 5;
  const bf16* arow  = A + (size_t)(m0 + l15)*Kc + half*16;
  const bf16* bbase = Bp + ((size_t)blockIdx.y*nKT*32 + lane)*16;
  v8f acc = {0.f,0.f,0.f,0.f,0.f,0.f,0.f,0.f};
  for(int kt=0; kt<nKT; ++kt){
    FragBF fa, fb;
    fa.q[0] = *reinterpret_cast<const v4f*>(arow + kt*32);
    fa.q[1] = *reinterpret_cast<const v4f*>(arow + kt*32 + 8);
    fb.q[0] = *reinterpret_cast<const v4f*>(bbase + (size_t)kt*512);
    fb.q[1] = *reinterpret_cast<const v4f*>(bbase + (size_t)kt*512 + 8);
    if(kt+1 < nKT){
      __builtin_prefetch(arow  + (kt+1)*32, 0, 1);        // global_prefetch_b8
      __builtin_prefetch(bbase + (size_t)(kt+1)*512, 0, 1);
    }
    acc = __builtin_amdgcn_wmma_f32_16x16x32_bf16(false, fa.v, false, fb.v,
                                                  (short)0, acc, false, false);
  }
  float scale = scaleP ? expf(*scaleP) : 1.0f;
  int n = n0 + l15;
  bool nOK = (n < NcOut);
  float bv = (bias && nOK) ? bias[n] : 0.0f;
  #pragma unroll
  for(int r=0;r<8;r++){
    int m = m0 + half*8 + r;
    float v = (acc[r] + bv)*scale;
    if(act==1) v = gelu_t(v);
    else if(act==2) v = v*sigm(v);
    else if(act==3) v = sigm(v);
    if(nOK){
      size_t o = (size_t)m*NcOut + n;
      if(outF) outF[o] = v;
      if(outB) outB[o] = (bf16)v;
    }
  }
}

// ---------------------------------------------------------------------------
// KNN: per residue i, 32 nearest is_aa + 16 nearest !is_aa (stable min select)
// ---------------------------------------------------------------------------
__global__ void knn_kernel(const float* __restrict__ pos, const float* __restrict__ am,
                           const unsigned char* __restrict__ isaa, int* __restrict__ nbr){
  __shared__ float da[N_RES];
  __shared__ float dsm[N_RES];
  __shared__ float rv[256];
  __shared__ int   ri[256];
  int i = blockIdx.x, t = threadIdx.x;
  float px = pos[(i*NATM+1)*3+0], py = pos[(i*NATM+1)*3+1], pz = pos[(i*NATM+1)*3+2];
  bool mi = am[i*NATM+1] > 0.f;
  for(int j=t;j<N_RES;j+=256){
    float dx=px-pos[(j*NATM+1)*3+0], dy=py-pos[(j*NATM+1)*3+1], dz=pz-pos[(j*NATM+1)*3+2];
    float d = sqrtf(dx*dx+dy*dy+dz*dz+1e-8f);
    bool ok = mi && (am[j*NATM+1] > 0.f);
    da[j]  = (ok &&  isaa[j]) ? d : BIGF;
    dsm[j] = (ok && !isaa[j]) ? d : BIGF;
  }
  __syncthreads();
  for(int slot=0;slot<KNB;slot++){
    float* arr = (slot<KAA)? da : dsm;
    float best=BIGF; int bi=N_RES;
    for(int j=t;j<N_RES;j+=256){ float v=arr[j]; if(v<best){best=v;bi=j;} }
    rv[t]=best; ri[t]=bi; __syncthreads();
    for(int s=128;s>0;s>>=1){
      if(t<s){ if(rv[t+s]<rv[t] || (rv[t+s]==rv[t] && ri[t+s]<ri[t])){ rv[t]=rv[t+s]; ri[t]=ri[t+s]; } }
      __syncthreads();
    }
    if(t==0){
      int sel = (rv[0] < 1e37f) ? ri[0] : -1;
      nbr[i*KNB+slot] = sel;
      if(sel>=0) arr[sel] = BIGF;
    }
    __syncthreads();
  }
}

// ---------------------------------------------------------------------------
// pair features (rbf16+3) @ W_pair_in (19x128) + LN  -- one wave per (n,k)
// ---------------------------------------------------------------------------
__global__ void pair_init_kernel(const float* __restrict__ pos, const int* __restrict__ nbr,
                                 const unsigned char* __restrict__ isaa,
                                 const int* __restrict__ chain, const int* __restrict__ resi,
                                 const float* __restrict__ Wpi,
                                 const float* __restrict__ lnS, const float* __restrict__ lnO,
                                 float* __restrict__ pairF, bf16* __restrict__ pairB){
  int wave = threadIdx.x>>5, lane = threadIdx.x&31;
  int p = blockIdx.x*8 + wave;
  if(p >= NKP) return;
  int n = p/KNB;
  int j = nbr[p];
  int jw = (j<0) ? N_RES-1 : j;       // jax gather wrap for -1
  float dx = pos[(n*NATM+1)*3+0]-pos[(jw*NATM+1)*3+0];
  float dy = pos[(n*NATM+1)*3+1]-pos[(jw*NATM+1)*3+1];
  float dz = pos[(n*NATM+1)*3+2]-pos[(jw*NATM+1)*3+2];
  float nd = sqrtf(dx*dx+dy*dy+dz*dz+1e-8f);
  float f[19];
  #pragma unroll
  for(int b=0;b<16;b++){
    float c = 2.0f + (20.0f/15.0f)*b;
    float z = (nd - c)/1.25f;
    f[b] = expf(-z*z);
  }
  f[16] = isaa[jw] ? 1.f : 0.f;
  bool sc = (chain[n]==chain[jw]);
  f[17] = (sc && resi[n]==resi[jw]) ? 1.f : 0.f;
  f[18] = (!sc) ? 1.f : 0.f;
  float o[4];
  #pragma unroll
  for(int q=0;q<4;q++){
    int c = lane + 32*q;
    float a = 0.f;
    for(int ff=0; ff<19; ++ff) a += f[ff]*Wpi[ff*DM + c];
    o[q]=a;
  }
  float m = waveSum(o[0]+o[1]+o[2]+o[3])*(1.0f/DM);
  float vv=0.f;
  #pragma unroll
  for(int q=0;q<4;q++){ float d0=o[q]-m; vv += d0*d0; }
  float inv = rsqrtf(waveSum(vv)*(1.0f/DM) + 1e-5f);
  #pragma unroll
  for(int q=0;q<4;q++){
    int c = lane + 32*q;
    float y = (o[q]-m)*inv*lnS[c] + lnO[c];
    pairF[(size_t)p*DM+c] = y;
    pairB[(size_t)p*DM+c] = (bf16)y;
  }
}

// ---------------------------------------------------------------------------
// local init: pair_weighted sum + [pw, is_aa, onehot21] @ W_local_in + LN
// ---------------------------------------------------------------------------
__global__ void local_init_kernel(const float* __restrict__ pwF, const int* __restrict__ nbr,
                                  const unsigned char* __restrict__ isaa, const int* __restrict__ aa,
                                  const float* __restrict__ Wli,
                                  const float* __restrict__ lnS, const float* __restrict__ lnO,
                                  float* __restrict__ localF, bf16* __restrict__ localB){
  __shared__ float pws[DM];
  __shared__ float red[DM];
  int n = blockIdx.x, t = threadIdx.x;
  float s = 0.f;
  for(int k=0;k<KNB;k++)
    if(nbr[n*KNB+k] != -1) s += pwF[((size_t)n*KNB+k)*DM + t];
  pws[t]=s; __syncthreads();
  float acc = 0.f;
  for(int ff=0; ff<DM; ++ff) acc += pws[ff]*Wli[ff*DM + t];
  acc += (isaa[n]?1.f:0.f)*Wli[DM*DM + t];
  acc += Wli[(DM+1+aa[n])*DM + t];
  float m = blkSum128(acc, red)*(1.f/DM);
  float d = acc - m;
  float var = blkSum128(d*d, red)*(1.f/DM);
  float y = d*rsqrtf(var+1e-5f)*lnS[t] + lnO[t];
  localF[(size_t)n*DM+t]=y; localB[(size_t)n*DM+t]=(bf16)y;
}

// ---------------------------------------------------------------------------
// feat = [local_i, local_j, pair]  (bf16)
// ---------------------------------------------------------------------------
__global__ void feat_kernel(const bf16* __restrict__ localB, const bf16* __restrict__ pairB,
                            const int* __restrict__ nbr, bf16* __restrict__ featB){
  size_t idx = (size_t)blockIdx.x*256 + threadIdx.x;
  if(idx >= (size_t)NKP*384) return;
  int c = (int)(idx % 384);
  size_t nk = idx / 384;
  int n = (int)(nk / KNB);
  bf16 v;
  if(c < DM)       v = localB[(size_t)n*DM + c];
  else if(c < 2*DM){
    int j = nbr[nk]; int jw = (j<0)? N_RES-1 : j;
    v = localB[(size_t)jw*DM + (c-DM)];
  } else           v = pairB[nk*DM + (c-2*DM)];
  featB[idx] = v;
}

// ---------------------------------------------------------------------------
// message epilogue: masked mean over K, sigmoid gate, residual + LN
// ---------------------------------------------------------------------------
__global__ void msg_epi_kernel(const float* __restrict__ uF, const int* __restrict__ nbr,
                               const float* __restrict__ am,
                               const float* __restrict__ gateW, const float* __restrict__ gateB,
                               const float* __restrict__ lnS, const float* __restrict__ lnO,
                               float* __restrict__ localF, bf16* __restrict__ localB){
  __shared__ float lrow[DM];
  __shared__ float red[DM];
  int n = blockIdx.x, t = threadIdx.x;
  lrow[t] = localF[(size_t)n*DM+t]; __syncthreads();
  bool mi = am[n*NATM+1] > 0.f;
  float u = 0.f;
  for(int k=0;k<KNB;k++)
    if(mi && nbr[n*KNB+k] != -1) u += uF[((size_t)n*KNB+k)*DM + t];
  u *= (1.0f/KNB);
  float g = 0.f;
  for(int f2=0; f2<DM; ++f2) g += lrow[f2]*gateW[f2*DM + t];
  g = sigm(g + gateB[t]);
  float x = lrow[t] + u*g;
  float m = blkSum128(x, red)*(1.f/DM);
  float d = x - m;
  float var = blkSum128(d*d, red)*(1.f/DM);
  float y = d*rsqrtf(var+1e-5f)*lnS[t] + lnO[t];
  localF[(size_t)n*DM+t]=y; localB[(size_t)n*DM+t]=(bf16)y;
}

__global__ void glu_mul_kernel(const bf16* __restrict__ g, const bf16* __restrict__ v,
                               bf16* __restrict__ gv){
  int idx = blockIdx.x*256 + threadIdx.x;
  if(idx >= N_RES*512) return;
  gv[idx] = (bf16)((float)g[idx]*(float)v[idx]);
}

__global__ void ln_res_kernel(const float* __restrict__ addF,
                              const float* __restrict__ lnS, const float* __restrict__ lnO,
                              float* __restrict__ localF, bf16* __restrict__ localB){
  __shared__ float red[DM];
  int n = blockIdx.x, t = threadIdx.x;
  size_t o = (size_t)n*DM + t;
  float x = localF[o] + addF[o];
  float m = blkSum128(x, red)*(1.f/DM);
  float d = x - m;
  float var = blkSum128(d*d, red)*(1.f/DM);
  float y = d*rsqrtf(var+1e-5f)*lnS[t] + lnO[t];
  localF[o]=y; localB[o]=(bf16)y;
}

__global__ void pair_epi_kernel(const float* __restrict__ uF, const float* __restrict__ gF,
                                const float* __restrict__ lnS, const float* __restrict__ lnO,
                                float* __restrict__ pairF, bf16* __restrict__ pairB){
  __shared__ float red[DM];
  int p = blockIdx.x, t = threadIdx.x;
  size_t o = (size_t)p*DM + t;
  float x = pairF[o] + uF[o]*gF[o];
  float m = blkSum128(x, red)*(1.f/DM);
  float d = x - m;
  float var = blkSum128(d*d, red)*(1.f/DM);
  float y = d*rsqrtf(var+1e-5f)*lnS[t] + lnO[t];
  pairF[o]=y; pairB[o]=(bf16)y;
}

// ---------------------------------------------------------------------------
// heads + Potts
// ---------------------------------------------------------------------------
__global__ void init_acc_kernel(float* acc){ if(threadIdx.x<16) acc[threadIdx.x]=0.f; }

__global__ void aa_nll_kernel(const float* __restrict__ aaL, const int* __restrict__ aagt,
                              const float* __restrict__ am, float* acc){
  int n = blockIdx.x*256 + threadIdx.x;
  if(n >= N_RES) return;
  float mx = -BIGF;
  for(int a=0;a<20;a++) mx = fmaxf(mx, aaL[n*20+a]);
  float se = 0.f;
  for(int a=0;a<20;a++) se += expf(aaL[n*20+a]-mx);
  if(am[n*NATM+1] > 0.f){
    atomicAdd(acc+0, (mx+logf(se)) - aaL[n*20+aagt[n]]);
    atomicAdd(acc+1, 1.0f);
  }
}

__global__ void aapair_nll_kernel(const bf16* __restrict__ ap, const int* __restrict__ nbr,
                                  const int* __restrict__ aagt, float* acc){
  __shared__ float v[400];
  __shared__ float red[128];
  int p = blockIdx.x, t = threadIdx.x;
  for(int e=t;e<400;e+=128) v[e] = (float)ap[(size_t)p*400+e];
  __syncthreads();
  float lm = -BIGF;
  for(int e=t;e<400;e+=128) lm = fmaxf(lm, v[e]);
  float mx = blkMax128(lm, red);
  float ls = 0.f;
  for(int e=t;e<400;e+=128) ls += expf(v[e]-mx);
  float se = blkSum128(ls, red);
  if(t==0){
    int j = nbr[p];
    if(j != -1){
      int n = p/KNB;
      float nll = (mx+logf(se)) - v[aagt[n]*20 + aagt[j]];
      atomicAdd(acc+2, nll); atomicAdd(acc+3, 1.0f);
    }
  }
}

__global__ void h_kernel(const float* __restrict__ pssmF, const float* __restrict__ aa_bias,
                         const float* __restrict__ am, float* __restrict__ hF){
  int n = blockIdx.x*256 + threadIdx.x;
  if(n >= N_RES) return;
  float m = 0.f;
  for(int a=0;a<20;a++) m += pssmF[n*20+a];
  m *= (1.f/20.f);
  bool mi = am[n*NATM+1] > 0.f;
  for(int a=0;a<20;a++)
    hF[n*20+a] = mi ? (pssmF[n*20+a] - m + aa_bias[a]) : 0.f;
}

__global__ void pottsA_kernel(const bf16* __restrict__ cl, const bf16* __restrict__ cr,
                              const int* __restrict__ nbr, const int* __restrict__ aagt,
                              const float* __restrict__ am,
                              float* __restrict__ Ja, float* __restrict__ Jb){
  __shared__ float clv[400];
  __shared__ float crv[400];
  int p = blockIdx.x, t = threadIdx.x;
  for(int e=t;e<400;e+=64){ clv[e]=(float)cl[(size_t)p*400+e]; crv[e]=(float)cr[(size_t)p*400+e]; }
  __syncthreads();
  int n = p/KNB, j = nbr[p];
  int jw = (j<0)? N_RES-1 : j;
  bool pm2 = (am[n*NATM+1]>0.f) && (am[jw*NATM+1]>0.f) && (j != -1);
  bool ns  = (j != n) && (j != -1);
  float fac = (pm2 && ns) ? 1.f : 0.f;
  int as = aagt[n], bs = aagt[jw];
  if(t < 20){
    float s=0.f;
    for(int c=0;c<20;c++) s += clv[t*20+c]*crv[c*20+bs];
    Ja[(size_t)p*20+t] = fac*s;
  } else if(t < 40){
    int b = t-20;
    float s=0.f;
    for(int c=0;c<20;c++) s += clv[as*20+c]*crv[c*20+b];
    Jb[(size_t)p*20+b] = fac*s;
  }
}

__global__ void ri_kernel(const float* __restrict__ hF, const float* __restrict__ Ja,
                          float* __restrict__ ri){
  int n = blockIdx.x, t = threadIdx.x;
  if(t < 20){
    float r = hF[n*20+t];
    for(int k=0;k<KNB;k++) r += Ja[((size_t)n*KNB+k)*20 + t];
    ri[n*20+t] = r;
  }
}

__global__ void pottsB_kernel(const bf16* __restrict__ cl, const bf16* __restrict__ cr,
                              const int* __restrict__ nbr, const int* __restrict__ aagt,
                              const float* __restrict__ am,
                              const float* __restrict__ Ja, const float* __restrict__ Jb,
                              const float* __restrict__ ri, float* acc){
  __shared__ float clv[400];
  __shared__ float crv[400];
  __shared__ float Mv[400];
  __shared__ float red[128];
  int p = blockIdx.x, t = threadIdx.x;
  for(int e=t;e<400;e+=128){ clv[e]=(float)cl[(size_t)p*400+e]; crv[e]=(float)cr[(size_t)p*400+e]; }
  __syncthreads();
  int n = p/KNB, j = nbr[p];
  int jw = (j<0)? N_RES-1 : j;
  bool pm2 = (am[n*NATM+1]>0.f) && (am[jw*NATM+1]>0.f) && (j != -1);
  bool ns  = (j != n) && (j != -1);
  float fac = (pm2 && ns) ? 1.f : 0.f;
  for(int e=t;e<400;e+=128){
    int a=e/20, b=e%20;
    float ct = 0.f;
    for(int c=0;c<20;c++) ct += clv[a*20+c]*crv[c*20+b];
    Mv[e] = -((ri[n*20+a]-Ja[(size_t)p*20+a]) + (ri[jw*20+b]-Jb[(size_t)p*20+b]) + fac*ct);
  }
  __syncthreads();
  float lm = -BIGF;
  for(int e=t;e<400;e+=128) lm = fmaxf(lm, Mv[e]);
  float mx = blkMax128(lm, red);
  float ls = 0.f;
  for(int e=t;e<400;e+=128) ls += expf(Mv[e]-mx);
  float se = blkSum128(ls, red);
  if(t==0 && pm2){
    float lp = Mv[aagt[n]*20 + aagt[jw]] - (mx + logf(se));
    atomicAdd(acc+4, lp); atomicAdd(acc+5, 1.f);
  }
}

__global__ void finalize_kernel(const float* acc, float* out){
  float aaN   =  acc[0]/fmaxf(1.f, acc[1]);
  float aaP   =  acc[2]/fmaxf(1.f, acc[3]);
  float potts = -acc[4]/fmaxf(acc[5], 1.f);
  out[0] = potts + aaN + aaP;
}

// ===========================================================================
// Host orchestration
// ===========================================================================
extern "C" void kernel_launch(void* const* d_in, const int* in_sizes, int n_in,
                              void* d_out, int out_size, void* d_ws, size_t ws_size,
                              hipStream_t stream){
  const float* pos  = (const float*)d_in[0];
  const float* am   = (const float*)d_in[1];
  const int*   aa   = (const int*)d_in[2];
  const int*   aagt = (const int*)d_in[3];
  const int*   chain= (const int*)d_in[4];
  const int*   resi = (const int*)d_in[5];
  const unsigned char* isaa = (const unsigned char*)d_in[6];

  // params flattened jax-style (sorted dict keys) starting at index 7
  const int P = 7;
  const float* E_scale = (const float*)d_in[P+0];
  const float* W_aa    = (const float*)d_in[P+1];
  const float* W_aap   = (const float*)d_in[P+2];
  const float* W_cl    = (const float*)d_in[P+3];
  const float* W_cr    = (const float*)d_in[P+4];
  const float* W_li    = (const float*)d_in[P+5];
  const float* W_pi    = (const float*)d_in[P+6];
  const float* W_pssm  = (const float*)d_in[P+7];
  const float* aa_bias = (const float*)d_in[P+8];
  const float* b_cl    = (const float*)d_in[P+9];
  const float* b_cr    = (const float*)d_in[P+10];
  struct Lp { const float *gate_W,*gate_b,*glu_Wg,*glu_Wo,*glu_Wv,*ln1_o,*ln1_s,*ln2_o,*ln2_s,
                          *lnp_o,*lnp_s,*msg_W1,*msg_W2,*pgate_W,*pgate_b,*pmsg_W1,*pmsg_W2; };
  Lp L[4];
  for(int l=0;l<4;l++){
    int b = P+11 + l*17;
    L[l].gate_W =(const float*)d_in[b+0];  L[l].gate_b =(const float*)d_in[b+1];
    L[l].glu_Wg =(const float*)d_in[b+2];  L[l].glu_Wo =(const float*)d_in[b+3];
    L[l].glu_Wv =(const float*)d_in[b+4];
    L[l].ln1_o  =(const float*)d_in[b+5];  L[l].ln1_s  =(const float*)d_in[b+6];
    L[l].ln2_o  =(const float*)d_in[b+7];  L[l].ln2_s  =(const float*)d_in[b+8];
    L[l].lnp_o  =(const float*)d_in[b+9];  L[l].lnp_s  =(const float*)d_in[b+10];
    L[l].msg_W1 =(const float*)d_in[b+11]; L[l].msg_W2 =(const float*)d_in[b+12];
    L[l].pgate_W=(const float*)d_in[b+13]; L[l].pgate_b=(const float*)d_in[b+14];
    L[l].pmsg_W1=(const float*)d_in[b+15]; L[l].pmsg_W2=(const float*)d_in[b+16];
  }
  int b2 = P+11+4*17;
  const float* ln_le_o = (const float*)d_in[b2+0];
  const float* ln_le_s = (const float*)d_in[b2+1];
  const float* ln_pe_o = (const float*)d_in[b2+2];
  const float* ln_pe_s = (const float*)d_in[b2+3];
  const float* peW1 = (const float*)d_in[b2+4];
  const float* peb1 = (const float*)d_in[b2+5];
  const float* peW2 = (const float*)d_in[b2+6];
  const float* peb2 = (const float*)d_in[b2+7];

  // ---- workspace bump allocator ----
  char* wsp = (char*)d_ws;
  size_t off = 0;
  auto alloc = [&](size_t bytes)->void*{
    void* p = wsp + off;
    off = (off + bytes + 255) & ~(size_t)255;
    return p;
  };
  int*   nbr    = (int*)  alloc((size_t)NKP*4);
  float* pairF  = (float*)alloc((size_t)NKP*DM*4);
  bf16*  pairB  = (bf16*) alloc((size_t)NKP*DM*2);
  float* localF = (float*)alloc((size_t)N_RES*DM*4);
  bf16*  localB = (bf16*) alloc((size_t)N_RES*DM*2);
  bf16*  featB  = (bf16*) alloc((size_t)NKP*384*2);
  bf16*  hidB   = (bf16*) alloc((size_t)NKP*512*2);
  float* uF     = (float*)alloc((size_t)NKP*DM*4);
  float* gateF  = (float*)alloc((size_t)NKP*DM*4);
  bf16*  gB     = (bf16*) alloc((size_t)N_RES*512*2);
  bf16*  vB     = (bf16*) alloc((size_t)N_RES*512*2);
  bf16*  gvB    = (bf16*) alloc((size_t)N_RES*512*2);
  float* ouF    = (float*)alloc((size_t)N_RES*DM*4);
  bf16*  bufP   = (bf16*) alloc((size_t)NKP*400*2);
  bf16*  bufQ   = (bf16*) alloc((size_t)NKP*400*2);
  float* aaLF   = (float*)alloc((size_t)N_RES*20*4);
  float* pssmF  = (float*)alloc((size_t)N_RES*20*4);
  float* hF     = (float*)alloc((size_t)N_RES*20*4);
  float* JaF    = (float*)alloc((size_t)NKP*20*4);
  float* JbF    = (float*)alloc((size_t)NKP*20*4);
  float* riF    = (float*)alloc((size_t)N_RES*20*4);
  float* accB   = (float*)alloc(64*4);
  // bf16 swizzled weights
  bf16* peW1B  = (bf16*)alloc((size_t)128*256*2);
  bf16* peW2B  = (bf16*)alloc((size_t)256*128*2);
  bf16* aaWB   = (bf16*)alloc((size_t)128*32*2);
  bf16* aapWB  = (bf16*)alloc((size_t)128*400*2);
  bf16* pssmWB = (bf16*)alloc((size_t)128*32*2);
  bf16* clWB   = (bf16*)alloc((size_t)128*400*2);
  bf16* crWB   = (bf16*)alloc((size_t)128*400*2);
  bf16 *msgW1B[4], *msgW2B[4], *gluWgB[4], *gluWvB[4], *gluWoB[4], *pmsgW1B[4], *pmsgW2B[4], *pgWB[4];
  for(int l=0;l<4;l++){
    msgW1B[l]  = (bf16*)alloc((size_t)384*512*2);
    msgW2B[l]  = (bf16*)alloc((size_t)512*128*2);
    gluWgB[l]  = (bf16*)alloc((size_t)128*512*2);
    gluWvB[l]  = (bf16*)alloc((size_t)128*512*2);
    gluWoB[l]  = (bf16*)alloc((size_t)512*128*2);
    pmsgW1B[l] = (bf16*)alloc((size_t)384*256*2);
    pmsgW2B[l] = (bf16*)alloc((size_t)256*128*2);
    pgWB[l]    = (bf16*)alloc((size_t)128*128*2);
  }

  auto conv = [&](const float* src, bf16* dst, int Kc, int Ncw, int NcPad){
    int total = Kc*NcPad;
    conv_weight_kernel<<<(total+255)/256, 256, 0, stream>>>(src, dst, Kc, Ncw, NcPad);
  };
  auto gemm = [&](const bf16* A, const bf16* Bp, float* oF, bf16* oB,
                  const float* bias, const float* scaleP,
                  int M, int Kc, int NcPad, int NcOut, int act){
    dim3 g(M/128, NcPad/16);
    gemm_wmma_kernel<<<g, 256, 0, stream>>>(A, Bp, oF, oB, bias, scaleP, M, Kc, NcOut, act);
  };

  // ---- weight conversion ----
  conv(peW1, peW1B, 128, 256, 256);
  conv(peW2, peW2B, 256, 128, 128);
  conv(W_aa,   aaWB,  128, 20, 32);
  conv(W_aap,  aapWB, 128, 400, 400);
  conv(W_pssm, pssmWB,128, 20, 32);
  conv(W_cl,   clWB,  128, 400, 400);
  conv(W_cr,   crWB,  128, 400, 400);
  for(int l=0;l<4;l++){
    conv(L[l].msg_W1,  msgW1B[l],  384, 512, 512);
    conv(L[l].msg_W2,  msgW2B[l],  512, 128, 128);
    conv(L[l].glu_Wg,  gluWgB[l],  128, 512, 512);
    conv(L[l].glu_Wv,  gluWvB[l],  128, 512, 512);
    conv(L[l].glu_Wo,  gluWoB[l],  512, 128, 128);
    conv(L[l].pmsg_W1, pmsgW1B[l], 384, 256, 256);
    conv(L[l].pmsg_W2, pmsgW2B[l], 256, 128, 128);
    conv(L[l].pgate_W, pgWB[l],    128, 128, 128);
  }

  init_acc_kernel<<<1, 32, 0, stream>>>(accB);

  // ---- graph construction + pair/local embeddings ----
  knn_kernel<<<N_RES, 256, 0, stream>>>(pos, am, isaa, nbr);
  pair_init_kernel<<<NKP/8, 256, 0, stream>>>(pos, nbr, isaa, chain, resi,
                                              W_pi, ln_pe_s, ln_pe_o, pairF, pairB);
  // pe MLP: gelu(pair@W1+b1)@W2 + b2
  gemm(pairB, peW1B, nullptr, hidB, peb1, nullptr, NKP, 128, 256, 256, 1);
  gemm(hidB,  peW2B, uF, nullptr, peb2, nullptr, NKP, 256, 128, 128, 0);
  local_init_kernel<<<N_RES, DM, 0, stream>>>(uF, nbr, isaa, aa, W_li, ln_le_s, ln_le_o,
                                              localF, localB);

  // ---- layers ----
  for(int l=0;l<4;l++){
    // node message
    feat_kernel<<<(unsigned)(((size_t)NKP*384)/256), 256, 0, stream>>>(localB, pairB, nbr, featB);
    gemm(featB, msgW1B[l], nullptr, hidB, nullptr, nullptr, NKP, 384, 512, 512, 1);
    gemm(hidB,  msgW2B[l], uF, nullptr, nullptr, nullptr, NKP, 512, 128, 128, 0);
    msg_epi_kernel<<<N_RES, DM, 0, stream>>>(uF, nbr, am, L[l].gate_W, L[l].gate_b,
                                             L[l].ln1_s, L[l].ln1_o, localF, localB);
    // GLU
    gemm(localB, gluWgB[l], nullptr, gB, nullptr, nullptr, N_RES, 128, 512, 512, 2); // silu
    gemm(localB, gluWvB[l], nullptr, vB, nullptr, nullptr, N_RES, 128, 512, 512, 0);
    glu_mul_kernel<<<(N_RES*512)/256, 256, 0, stream>>>(gB, vB, gvB);
    gemm(gvB, gluWoB[l], ouF, nullptr, nullptr, nullptr, N_RES, 512, 128, 128, 0);
    ln_res_kernel<<<N_RES, DM, 0, stream>>>(ouF, L[l].ln2_s, L[l].ln2_o, localF, localB);
    // pair message (uses updated local, pre-update pair)
    feat_kernel<<<(unsigned)(((size_t)NKP*384)/256), 256, 0, stream>>>(localB, pairB, nbr, featB);
    gemm(featB, pmsgW1B[l], nullptr, hidB, nullptr, nullptr, NKP, 384, 256, 256, 1);
    gemm(hidB,  pmsgW2B[l], uF, nullptr, nullptr, nullptr, NKP, 256, 128, 128, 0);
    gemm(pairB, pgWB[l], gateF, nullptr, L[l].pgate_b, nullptr, NKP, 128, 128, 128, 3); // sigmoid
    pair_epi_kernel<<<NKP, DM, 0, stream>>>(uF, gateF, L[l].lnp_s, L[l].lnp_o, pairF, pairB);
  }

  // ---- heads ----
  gemm(localB, aaWB, aaLF, nullptr, nullptr, nullptr, N_RES, 128, 32, 20, 0);
  aa_nll_kernel<<<(N_RES+255)/256, 256, 0, stream>>>(aaLF, aagt, am, accB);

  gemm(pairB, aapWB, nullptr, bufP, nullptr, nullptr, NKP, 128, 400, 400, 0);
  aapair_nll_kernel<<<NKP, 128, 0, stream>>>(bufP, nbr, aagt, accB);

  gemm(localB, pssmWB, pssmF, nullptr, nullptr, E_scale, N_RES, 128, 32, 20, 0);
  h_kernel<<<(N_RES+255)/256, 256, 0, stream>>>(pssmF, aa_bias, am, hF);

  gemm(pairB, clWB, nullptr, bufP, b_cl, E_scale, NKP, 128, 400, 400, 0);
  gemm(pairB, crWB, nullptr, bufQ, b_cr, E_scale, NKP, 128, 400, 400, 0);

  pottsA_kernel<<<NKP, 64, 0, stream>>>(bufP, bufQ, nbr, aagt, am, JaF, JbF);
  ri_kernel<<<N_RES, 32, 0, stream>>>(hF, JaF, riF);
  pottsB_kernel<<<NKP, 128, 0, stream>>>(bufP, bufQ, nbr, aagt, am, JaF, JbF, riF, accB);

  finalize_kernel<<<1, 1, 0, stream>>>(accB, (float*)d_out);
}
